// Model_87316685127944
// MI455X (gfx1250) — compile-verified, matrix operand order checked by reading, not run on hardware
//
#include <hip/hip_runtime.h>
#include <hip/hip_bf16.h>
#include <stdint.h>

typedef unsigned short u16;
typedef unsigned int u32;
typedef __attribute__((ext_vector_type(16))) __bf16 v16bf;
typedef __attribute__((ext_vector_type(8)))  float  v8f;

#define D_MODEL  1024
#define N_HEADS  16
#define HEAD_DIM 64
#define SCS      2064   // score row stride in floats (2048 + pad)

#if defined(__HIP_DEVICE_COMPILE__) && __has_builtin(__builtin_amdgcn_global_load_async_to_lds_b64)
#define HAS_ASYNC_LDS 1
#else
#define HAS_ASYNC_LDS 0
#endif

// fp32 -> bf16 round-to-nearest-even
__device__ __forceinline__ u16 f2bf(float x) {
  u32 u = __float_as_uint(x);
  u32 r = (u + 0x7FFFu + ((u >> 16) & 1u)) >> 16;
  return (u16)r;
}

union FragB16 {
  v16bf v;
  u32 u[8];
  u16 s[16];
};

// 8-byte global -> LDS copy; async (GLOBAL_LOAD_ASYNC_TO_LDS_B64) when available.
typedef int v2i_vs __attribute__((__vector_size__(2 * sizeof(int))));
typedef __attribute__((address_space(1))) v2i_vs gv2i;
typedef __attribute__((address_space(3))) v2i_vs lv2i;

__device__ __forceinline__ void cp_g2l_b64(const u16* __restrict__ g, u16* l) {
#if HAS_ASYNC_LDS
  __builtin_amdgcn_global_load_async_to_lds_b64((gv2i*)g, (lv2i*)l, 0, 0);
#else
  const u32* s = (const u32*)g;
  u32* d = (u32*)l;
  d[0] = s[0];
  d[1] = s[1];
#endif
}

__device__ __forceinline__ void cp_g2l_wait() {
#if HAS_ASYNC_LDS
#if __has_builtin(__builtin_amdgcn_s_wait_asynccnt)
  __builtin_amdgcn_s_wait_asynccnt(0);
#else
  asm volatile("s_wait_asynccnt 0" ::: "memory");
#endif
#endif
}

// Load a 16x32 bf16 A-layout fragment (also used for B as A-layout of B^T).
// ISA 7.12.2: lanes 0-15 / 16-31 take the two K-halves; VGPR v<4 holds
// K = kh+2v,kh+2v+1 ; VGPR v>=4 holds K = 16+kh+2(v-4),...
__device__ __forceinline__ v16bf load_frag_b16(const u16* buf, int stride, int row, int k0) {
  int lane = threadIdx.x & 31;
  int r  = row + (lane & 15);
  int kh = (lane >> 4) * 8;
  const u32* p0 = (const u32*)(buf + r * stride + k0 + kh);
  const u32* p1 = (const u32*)(buf + r * stride + k0 + kh + 16);
  FragB16 f;
#pragma unroll
  for (int i = 0; i < 4; ++i) { f.u[i] = p0[i]; f.u[4 + i] = p1[i]; }
  return f.v;
}

// C[M,N] = A[M,K] * W[K,N] + bias ; A fp32 or bf16, out bf16 head-split or fp32.
// Block tile 128x64, K-step 32, 8 waves (wave32) each computing 32x32.
template<bool A_BF16, bool SPLIT_HEADS>
__global__ __launch_bounds__(256) void gemm_bf16(
    const void* __restrict__ Aptr, const float* __restrict__ W,
    const float* __restrict__ bias, void* __restrict__ Cptr,
    int M, int K, int N, int L)
{
  __shared__ u16 As[128 * 32];
  __shared__ u16 Bs[64 * 32];   // W^T tile: Bs[n][k]
  const int t    = threadIdx.x;
  const int m0   = blockIdx.y * 128;
  const int n0   = blockIdx.x * 64;
  const int w    = t >> 5;
  const int lane = t & 31;
  const int wm   = (w & 3) * 32;
  const int wn   = (w >> 2) * 32;

  v8f acc[2][2] = {};

  for (int k0 = 0; k0 < K; k0 += 32) {
    __syncthreads();
    // --- stage A tile (128 x 32) as bf16 ---
    {
      int ar = t >> 3;
      int ac = (t & 7) * 4;
      if (A_BF16) {
#pragma unroll
        for (int p = 0; p < 4; ++p) {
          int row = p * 32 + ar;
          cp_g2l_b64((const u16*)Aptr + (size_t)(m0 + row) * K + k0 + ac,
                     &As[row * 32 + ac]);
        }
      } else {
        float4 fa[4];
#pragma unroll
        for (int p = 0; p < 4; ++p)
          fa[p] = *(const float4*)((const float*)Aptr + (size_t)(m0 + p * 32 + ar) * K + k0 + ac);
#pragma unroll
        for (int p = 0; p < 4; ++p) {
          u32* dst = (u32*)&As[(p * 32 + ar) * 32 + ac];
          dst[0] = (u32)f2bf(fa[p].x) | ((u32)f2bf(fa[p].y) << 16);
          dst[1] = (u32)f2bf(fa[p].z) | ((u32)f2bf(fa[p].w) << 16);
        }
      }
    }
    // --- stage W tile (32 x 64) transposed -> Bs[n][k], converted to bf16 ---
    {
      int wr = t >> 4;
      int wc = (t & 15) * 4;
      float4 fw[2];
#pragma unroll
      for (int p = 0; p < 2; ++p)
        fw[p] = *(const float4*)(W + (size_t)(k0 + p * 16 + wr) * N + n0 + wc);
#pragma unroll
      for (int p = 0; p < 2; ++p) {
        int kk = p * 16 + wr;
        Bs[(wc + 0) * 32 + kk] = f2bf(fw[p].x);
        Bs[(wc + 1) * 32 + kk] = f2bf(fw[p].y);
        Bs[(wc + 2) * 32 + kk] = f2bf(fw[p].z);
        Bs[(wc + 3) * 32 + kk] = f2bf(fw[p].w);
      }
    }
    cp_g2l_wait();
    __syncthreads();
    v16bf a0 = load_frag_b16(As, 32, wm, 0);
    v16bf a1 = load_frag_b16(As, 32, wm + 16, 0);
    v16bf b0 = load_frag_b16(Bs, 32, wn, 0);
    v16bf b1 = load_frag_b16(Bs, 32, wn + 16, 0);
    acc[0][0] = __builtin_amdgcn_wmma_f32_16x16x32_bf16(false, a0, false, b0, (short)0, acc[0][0], false, false);
    acc[0][1] = __builtin_amdgcn_wmma_f32_16x16x32_bf16(false, a0, false, b1, (short)0, acc[0][1], false, false);
    acc[1][0] = __builtin_amdgcn_wmma_f32_16x16x32_bf16(false, a1, false, b0, (short)0, acc[1][0], false, false);
    acc[1][1] = __builtin_amdgcn_wmma_f32_16x16x32_bf16(false, a1, false, b1, (short)0, acc[1][1], false, false);
  }

  // epilogue: C/D layout -> lane holds N = lane&15, VGPR i -> M = (lane>>4)*8 + i
  const int Nl = lane & 15;
  const int Mb = (lane >> 4) * 8;
#pragma unroll
  for (int ti = 0; ti < 2; ++ti)
#pragma unroll
    for (int tj = 0; tj < 2; ++tj)
#pragma unroll
      for (int i = 0; i < 8; ++i) {
        int gm = m0 + wm + ti * 16 + Mb + i;
        int gn = n0 + wn + tj * 16 + Nl;
        float val = acc[ti][tj][i] + bias[gn];
        if (SPLIT_HEADS) {
          int b = gm / L, l = gm % L;
          int h = gn >> 6, d = gn & 63;
          ((u16*)Cptr)[(((size_t)(b * N_HEADS + h) * L + l) << 6) + d] = f2bf(val);
        } else {
          ((float*)Cptr)[(size_t)gm * N + gn] = val;
        }
      }
}

// One block = one (b,h) x 16 query rows. Scores strip lives in LDS.
// sparsemax tau via bisection on g(tau) = sum relu(s - tau) - 1 (monotone).
__global__ __launch_bounds__(128) void attn_sparsemax(
    const u16* __restrict__ Qh, const u16* __restrict__ Kh,
    const u16* __restrict__ Vh, u16* __restrict__ ctx, int L)
{
  extern __shared__ char smem[];
  float* sc  = (float*)smem;                 // 16 x SCS fp32 scores
  u16*   Ks  = (u16*)(sc + 16 * SCS);        // 64 x 80 key block / V^T block
  u16*   Qs  = Ks + 64 * 80;                 // 16 x 80
  float* tau = (float*)(Qs + 16 * 80);       // 16

  const int t    = threadIdx.x;
  const int q0   = blockIdx.x * 16;
  const int bh   = blockIdx.y;
  const size_t base = (size_t)bh * L * HEAD_DIM;
  const int w    = t >> 5;
  const int lane = t & 31;
  const int Nl   = lane & 15;
  const int Mb   = (lane >> 4) * 8;

  // load Q tile 16x64 bf16 (async global->LDS)
#pragma unroll
  for (int p = 0; p < 2; ++p) {
    int chunk = p * 128 + t;
    int r  = chunk >> 4;
    int c4 = (chunk & 15) * 4;
    cp_g2l_b64(Qh + base + (size_t)(q0 + r) * HEAD_DIM + c4, &Qs[r * 80 + c4]);
  }
  cp_g2l_wait();
  __syncthreads();

  const float scale = 0.125f;  // 1/sqrt(64)

  // phase 1: scores = scale * Q K^T  (per 64-key block, 4 waves x 16 cols)
  for (int kb = 0; kb < L / 64; ++kb) {
    __syncthreads();
#pragma unroll
    for (int p = 0; p < 8; ++p) {
      int chunk = p * 128 + t;
      int kk = chunk >> 4;
      int c4 = (chunk & 15) * 4;
      cp_g2l_b64(Kh + base + (size_t)(kb * 64 + kk) * HEAD_DIM + c4, &Ks[kk * 80 + c4]);
    }
    cp_g2l_wait();
    __syncthreads();
    v8f s = {};
#pragma unroll
    for (int ks = 0; ks < 2; ++ks) {
      v16bf a = load_frag_b16(Qs, 80, 0, ks * 32);
      v16bf b = load_frag_b16(Ks, 80, w * 16, ks * 32);
      s = __builtin_amdgcn_wmma_f32_16x16x32_bf16(false, a, false, b, (short)0, s, false, false);
    }
#pragma unroll
    for (int i = 0; i < 8; ++i)
      sc[(Mb + i) * SCS + kb * 64 + w * 16 + Nl] = s[i] * scale;
  }
  __syncthreads();

  // phase 2: sparsemax threshold (8 lanes per row, butterfly reductions)
  {
    int row = t >> 3, sub = t & 7;
    const float* r = sc + row * SCS;
    float m = -3.4e38f;
    for (int c = sub; c < L; c += 8) m = fmaxf(m, r[c]);
#pragma unroll
    for (int o = 4; o; o >>= 1) m = fmaxf(m, __shfl_xor(m, o, 32));
    float lo = m - 1.0f, hi = m;  // tau in [max-1, max]
#pragma unroll 1
    for (int it = 0; it < 32; ++it) {
      float tm = 0.5f * (lo + hi);
      float ssum = 0.0f;
      for (int c = sub; c < L; c += 8) ssum += fmaxf(r[c] - tm, 0.0f);
#pragma unroll
      for (int o = 4; o; o >>= 1) ssum += __shfl_xor(ssum, o, 32);
      if (ssum > 1.0f) lo = tm; else hi = tm;
    }
    if (sub == 0) tau[row] = 0.5f * (lo + hi);
  }
  __syncthreads();

  // phase 3: out = relu(S - tau) @ V   (wave w owns d-cols w*16..w*16+15)
  v8f o = {};
  const int lr = lane & 15;
  const int kh = (lane >> 4) * 8;
  const float tr = tau[lr];
  for (int kb = 0; kb < L / 64; ++kb) {
    __syncthreads();
    // stage V block transposed: Ks[d][kk]  (B fragment = A-layout of V^T)
    {
      u32 w0[8], w1[8];
#pragma unroll
      for (int p = 0; p < 8; ++p) {
        int chunk = p * 128 + t;
        int kk = chunk >> 4;
        int c4 = (chunk & 15) * 4;
        const u32* src = (const u32*)(Vh + base + (size_t)(kb * 64 + kk) * HEAD_DIM + c4);
        w0[p] = src[0];
        w1[p] = src[1];
      }
#pragma unroll
      for (int p = 0; p < 8; ++p) {
        int chunk = p * 128 + t;
        int kk = chunk >> 4;
        int c4 = (chunk & 15) * 4;
        Ks[(c4 + 0) * 80 + kk] = (u16)(w0[p] & 0xffffu);
        Ks[(c4 + 1) * 80 + kk] = (u16)(w0[p] >> 16);
        Ks[(c4 + 2) * 80 + kk] = (u16)(w1[p] & 0xffffu);
        Ks[(c4 + 3) * 80 + kk] = (u16)(w1[p] >> 16);
      }
    }
    __syncthreads();
#pragma unroll
    for (int ks = 0; ks < 2; ++ks) {
      FragB16 fa;
      const float* rr = sc + lr * SCS + kb * 64 + ks * 32 + kh;
#pragma unroll
      for (int j = 0; j < 8; ++j) {
        fa.s[j]     = f2bf(fmaxf(rr[j]      - tr, 0.0f));
        fa.s[8 + j] = f2bf(fmaxf(rr[16 + j] - tr, 0.0f));
      }
      v16bf vb = load_frag_b16(Ks, 80, w * 16, ks * 32);
      o = __builtin_amdgcn_wmma_f32_16x16x32_bf16(false, fa.v, false, vb, (short)0, o, false, false);
    }
  }

  // write context [B, L, D] bf16 (merged heads)
  const int b = bh >> 4, h = bh & 15;
  const int d = w * 16 + Nl;
#pragma unroll
  for (int i = 0; i < 8; ++i) {
    int gq = q0 + Mb + i;
    ctx[(size_t)(b * L + gq) * D_MODEL + h * HEAD_DIM + d] = f2bf(o[i]);
  }
}

extern "C" void kernel_launch(void* const* d_in, const int* in_sizes, int n_in,
                              void* d_out, int out_size, void* d_ws, size_t ws_size,
                              hipStream_t stream) {
  (void)in_sizes; (void)n_in; (void)out_size; (void)ws_size;
  const float* q  = (const float*)d_in[0];
  const float* k  = (const float*)d_in[1];
  const float* v  = (const float*)d_in[2];
  const float* Wq = (const float*)d_in[3];
  const float* bq = (const float*)d_in[4];
  const float* Wk = (const float*)d_in[5];
  const float* bk = (const float*)d_in[6];
  const float* Wv = (const float*)d_in[7];
  const float* bv = (const float*)d_in[8];
  const float* Wo = (const float*)d_in[9];
  const float* bo = (const float*)d_in[10];

  const int B = 2, L = 2048, M = B * L;  // 4096 rows
  u16* Qh  = (u16*)d_ws;                       // [B,H,L,64] bf16, 8 MB
  u16* Kh  = Qh + (size_t)M * D_MODEL;
  u16* Vh  = Kh + (size_t)M * D_MODEL;
  u16* ctx = Vh + (size_t)M * D_MODEL;         // [B,L,D] bf16

  dim3 gg(D_MODEL / 64, M / 128);  // (16, 32)
  gemm_bf16<false, true><<<gg, 256, 0, stream>>>(q, Wq, bq, Qh, M, D_MODEL, D_MODEL, L);
  gemm_bf16<false, true><<<gg, 256, 0, stream>>>(k, Wk, bk, Kh, M, D_MODEL, D_MODEL, L);
  gemm_bf16<false, true><<<gg, 256, 0, stream>>>(v, Wv, bv, Vh, M, D_MODEL, D_MODEL, L);

  size_t shmem = (size_t)16 * SCS * 4 + 64 * 80 * 2 + 16 * 80 * 2 + 16 * 4;  // ~145 KB (<320 KB/WGP)
  attn_sparsemax<<<dim3(L / 16, B * N_HEADS), 128, shmem, stream>>>(Qh, Kh, Vh, ctx, L);

  gemm_bf16<true, false><<<gg, 256, 0, stream>>>(ctx, Wo, bo, d_out, M, D_MODEL, D_MODEL, L);
}
